// GC_GRU_57088705298508
// MI455X (gfx1250) — matrix operand order repaired and checked by reading, not compiled, not run on hardware
//
#include <hip/hip_runtime.h>
#include <cstdint>
#include <cstddef>

// ---------------- problem constants ----------------
#define BB      32
#define NNODES  1500
#define THIST   24
#define TPRED   24
#define FDIMC   31
#define IND     32          // OUT + FDIM
#define HID     256
#define GCNF    256
#define EDG     24000
#define MROWS   (BB * NNODES)   // 48000
#define XGK     (IND + GCNF)    // 288
#define GATES   (3 * HID)       // 768
#define CHK     (3 * IND)       // 96
#define TTOT    (THIST + TPRED) // 48

// ---------------- helpers ----------------
__device__ __forceinline__ unsigned short f2bf(float f) {
  unsigned int u = __float_as_uint(f);
  unsigned int r = (u + 0x7FFFu + ((u >> 16) & 1u)) >> 16;   // RNE
  return (unsigned short)r;
}
__device__ __forceinline__ float sigm(float x) { return 1.0f / (1.0f + __expf(-x)); }

typedef __attribute__((ext_vector_type(16))) __bf16 v16bf;
typedef __attribute__((ext_vector_type(8)))  float  v8f;

struct U4x2 { uint4 a, b; };
// build a 16-element bf16 fragment from two contiguous 16-byte LDS chunks
__device__ __forceinline__ v16bf frag16(const uint32_t* p0, const uint32_t* p1) {
  U4x2 t;
  t.a = *(const uint4*)p0;
  t.b = *(const uint4*)p1;
  return __builtin_bit_cast(v16bf, t);
}

// ---- optional CDNA5 async global->LDS path (probe via __has_builtin) ----
#if defined(__AMDGCN__) && __has_builtin(__builtin_amdgcn_global_load_async_to_lds_b128) && \
    __has_builtin(__builtin_amdgcn_s_wait_asynccnt)
#define USE_ASYNC_LDS 1
typedef int v4i_vs __attribute__((__vector_size__(4 * sizeof(int))));
typedef v4i_vs __attribute__((address_space(1)))* gptr_v4i;
typedef v4i_vs __attribute__((address_space(3)))* lptr_v4i;
__device__ __forceinline__ gptr_v4i to_glob(const void* p) {
  return (gptr_v4i)(uintptr_t)p;
}
__device__ __forceinline__ lptr_v4i to_lds(void* p) {
  // generic LDS address truncates to the 32-bit LDS offset (aperture scheme)
  return (lptr_v4i)(uint32_t)(uintptr_t)p;
}
#else
#define USE_ASYNC_LDS 0
#endif

// ---------------- graph preprocessing ----------------
__global__ void k_deg(const int* __restrict__ src, float* __restrict__ deg) {
  int e = blockIdx.x * 256 + threadIdx.x;
  if (e < EDG) atomicAdd(&deg[src[e]], 1.0f);
}

__global__ void k_edgew(const int* __restrict__ src, const int* __restrict__ dst,
                        const float* __restrict__ deg, float* __restrict__ w) {
  int e = blockIdx.x * 256 + threadIdx.x;
  if (e >= EDG) return;
  float a = deg[src[e]], b = deg[dst[e]];
  float da = (a > 0.0f) ? rsqrtf(fmaxf(a, 1.0f)) : 0.0f;
  float db = (b > 0.0f) ? rsqrtf(fmaxf(b, 1.0f)) : 0.0f;
  w[e] = -(da * db);
}

// ---------------- weight conversion ----------------
__global__ void k_f2bf(const float* __restrict__ x, unsigned short* __restrict__ y, int n) {
  int i = blockIdx.x * 256 + threadIdx.x;
  if (i < n) y[i] = f2bf(x[i]);
}
// W [rows x cols] fp32 -> WT [cols x rows] bf16  (used for W_cheb [K x N] -> [N x K])
__global__ void k_transpose_bf(const float* __restrict__ W, unsigned short* __restrict__ WT,
                               int rows, int cols) {
  int i = blockIdx.x * 256 + threadIdx.x;
  if (i >= rows * cols) return;
  int r = i / cols, c = i % cols;
  WT[(size_t)c * rows + r] = f2bf(W[i]);
}

// ---------------- per-step kernels ----------------
__global__ void k_build_xt(const float* __restrict__ xhist, const float* __restrict__ xprev,
                           int use_hist, int thist, const float* __restrict__ feats, int tf,
                           float* __restrict__ Tx0, unsigned short* __restrict__ xg,
                           unsigned short* __restrict__ Acheb) {
  int i = blockIdx.x * 256 + threadIdx.x;          // m*32 + d
  if (i >= MROWS * IND) return;
  int m = i >> 5, d = i & 31;
  int b = m / NNODES, n = m % NNODES;
  float v;
  if (d == 0) {
    v = use_hist ? xhist[((size_t)b * THIST + thist) * NNODES + n] : xprev[m];
  } else {
    v = feats[(((size_t)b * TTOT + tf) * NNODES + n) * FDIMC + (d - 1)];
  }
  Tx0[(size_t)m * IND + d] = v;
  unsigned short bv = f2bf(v);
  xg[(size_t)m * XGK + d] = bv;
  Acheb[(size_t)m * CHK + d] = bv;
}

// Tout[:, dst, :] += w[e] * Tin[:, src, :]
__global__ void k_scatter(const float* __restrict__ Tin, float* __restrict__ Tout,
                          const int* __restrict__ src, const int* __restrict__ dst,
                          const float* __restrict__ w) {
  int idx = blockIdx.x * 256 + threadIdx.x;        // b*EDG + e
  if (idx >= BB * EDG) return;
  int b = idx / EDG, e = idx % EDG;
  float wv = w[e];
  const float* s = Tin  + ((size_t)b * NNODES + src[e]) * IND;
  float*       o = Tout + ((size_t)b * NNODES + dst[e]) * IND;
  #pragma unroll
  for (int d = 0; d < IND; ++d) atomicAdd(&o[d], wv * s[d]);
}

// Acheb[:,32:64] = bf16(Tx1) ; Acheb[:,64:96] = bf16(2*Tx2raw - Tx0)
__global__ void k_combine(const float* __restrict__ Tx0, const float* __restrict__ Tx1,
                          const float* __restrict__ Tx2raw, unsigned short* __restrict__ Acheb) {
  int i = blockIdx.x * 256 + threadIdx.x;
  if (i >= MROWS * IND) return;
  int m = i >> 5, d = i & 31;
  Acheb[(size_t)m * CHK + 32 + d] = f2bf(Tx1[i]);
  Acheb[(size_t)m * CHK + 64 + d] = f2bf(2.0f * Tx2raw[i] - Tx0[i]);
}

// ---------------- WMMA GEMM:  C[M x Nn] = A[M x K](bf16) * Bt[Nn x K](bf16, N-major) --------
// MODE 0: C[m][n] = acc + bias[n]                      (fp32 out)
// MODE 1: Obf[m*ldo + ocol + n] = bf16(sigmoid(acc+b)) (fused Cheb epilogue -> xg)
template <int MODE>
__global__ __launch_bounds__(128) void k_gemm_bf16(
    const unsigned short* __restrict__ A, const unsigned short* __restrict__ Bt,
    int Nn, int K, const float* __restrict__ bias,
    float* __restrict__ C, unsigned short* __restrict__ Obf, int ldo, int ocol) {
  __shared__ uint32_t As32[64 * 16];   // 64 rows (M) x 32 bf16 (K)
  __shared__ uint32_t Bs32[64 * 16];   // 64 rows (N) x 32 bf16 (K)

  const int tid  = threadIdx.x;
  const int lane = tid & 31;
  const int wave = tid >> 5;
  const int wm0  = (wave >> 1) * 32;   // wave's 32x32 patch in 64x64 block tile
  const int wn0  = (wave & 1) * 32;
  const int bm   = blockIdx.x * 64;
  const int bn   = blockIdx.y * 64;

  v8f cc[2][2];
  cc[0][0] = v8f{}; cc[0][1] = v8f{}; cc[1][0] = v8f{}; cc[1][1] = v8f{};

  const uint32_t* Ap = (const uint32_t*)A;
  const uint32_t* Bp = (const uint32_t*)Bt;
  const int ldku = K >> 1;             // uints per row for both operands
  const int kTiles = K >> 5;

  for (int kt = 0; kt < kTiles; ++kt) {
    const int kh = kt << 4;            // uint offset of this 32-wide K slice
#if USE_ASYNC_LDS
    #pragma unroll
    for (int i = 0; i < 2; ++i) {
      int chunk = tid + i * 128;       // 0..255, 16B each
      int r = chunk >> 2, c4 = (chunk & 3) << 2;
      __builtin_amdgcn_global_load_async_to_lds_b128(
          to_glob(Ap + (size_t)(bm + r) * ldku + kh + c4), to_lds(As32 + chunk * 4), 0, 0);
      __builtin_amdgcn_global_load_async_to_lds_b128(
          to_glob(Bp + (size_t)(bn + r) * ldku + kh + c4), to_lds(Bs32 + chunk * 4), 0, 0);
    }
    __builtin_amdgcn_s_wait_asynccnt(0);
#else
    #pragma unroll
    for (int i = 0; i < 8; ++i) {
      int idx = tid + i * 128;         // 0..1023
      int r = idx >> 4, c = idx & 15;
      As32[idx] = Ap[(size_t)(bm + r) * ldku + kh + c];
      Bs32[idx] = Bp[(size_t)(bn + r) * ldku + kh + c];
    }
#endif
    __syncthreads();

    // ---- fragments per documented CDNA5 16-bit layouts (all b128 LDS reads) ----
    const uint32_t* Ar0 = As32 + (wm0 + (lane & 15)) * 16;
    const uint32_t* Br0 = Bs32 + (wn0 + (lane & 15)) * 16;
    const int kb2a = (lane >> 4) * 4;  // A: K pairs (0..7 | 8..15)/2
    const int kb2b = (lane >> 4) * 4;  // B: K pairs (0..7 | 8..15)/2
    v16bf a0 = frag16(Ar0 + kb2a,       Ar0 + kb2a + 8);
    v16bf a1 = frag16(Ar0 + 256 + kb2a, Ar0 + 256 + kb2a + 8);   // +16 rows * 16 uints
    v16bf b0 = frag16(Br0 + kb2b,       Br0 + kb2b + 8);
    v16bf b1 = frag16(Br0 + 256 + kb2b, Br0 + 256 + kb2b + 8);

    cc[0][0] = __builtin_amdgcn_wmma_f32_16x16x32_bf16(false, a0, false, b0, (short)0, cc[0][0], false, false);
    cc[0][1] = __builtin_amdgcn_wmma_f32_16x16x32_bf16(false, a0, false, b1, (short)0, cc[0][1], false, false);
    cc[1][0] = __builtin_amdgcn_wmma_f32_16x16x32_bf16(false, a1, false, b0, (short)0, cc[1][0], false, false);
    cc[1][1] = __builtin_amdgcn_wmma_f32_16x16x32_bf16(false, a1, false, b1, (short)0, cc[1][1], false, false);
    __syncthreads();
  }

  // ---- epilogue (C layout: VGPR j -> M=j / M=8+j per lane-half; N = lane&15) ----
  const int rsel = (lane >> 4) * 8;
  const int csel = lane & 15;
  float bco[2];
  #pragma unroll
  for (int tn = 0; tn < 2; ++tn) bco[tn] = bias[bn + wn0 + tn * 16 + csel];

  #pragma unroll
  for (int tm = 0; tm < 2; ++tm) {
    #pragma unroll
    for (int tn = 0; tn < 2; ++tn) {
      #pragma unroll
      for (int j = 0; j < 8; ++j) {
        int gm = bm + wm0 + tm * 16 + rsel + j;
        int gn = bn + wn0 + tn * 16 + csel;
        float v = cc[tm][tn][j] + bco[tn];
        if (MODE == 0) {
          C[(size_t)gm * Nn + gn] = v;
        } else {
          Obf[(size_t)gm * ldo + ocol + gn] = f2bf(sigm(v));
        }
      }
    }
  }
}

// ---------------- GRU pointwise ----------------
__global__ void k_gru(const float* __restrict__ gi, const float* __restrict__ gh,
                      float* __restrict__ h, unsigned short* __restrict__ hbf) {
  size_t i = (size_t)blockIdx.x * 256 + threadIdx.x;   // m*256 + j
  if (i >= (size_t)MROWS * HID) return;
  size_t m = i >> 8; int j = (int)(i & 255);
  size_t gb = m * GATES;
  float r  = sigm(gi[gb + j]       + gh[gb + j]);
  float z  = sigm(gi[gb + 256 + j] + gh[gb + 256 + j]);
  float nn = tanhf(gi[gb + 512 + j] + r * gh[gb + 512 + j]);
  float hv = h[i];
  float hn = (1.0f - z) * nn + z * hv;
  h[i] = hn;
  hbf[i] = f2bf(hn);
}

// ---------------- decoder FC + output ----------------
__global__ void k_fc(const float* __restrict__ h, const float* __restrict__ Wfc,
                     const float* __restrict__ bfc, float* __restrict__ xprev,
                     float* __restrict__ out, int s) {
  int m = blockIdx.x * 256 + threadIdx.x;
  if (m >= MROWS) return;
  const float* hr = h + (size_t)m * HID;
  float acc = 0.0f;
  #pragma unroll 8
  for (int j = 0; j < HID; ++j) acc += hr[j] * Wfc[j];
  acc += bfc[0];
  xprev[m] = acc;
  int b = m / NNODES, n = m % NNODES;
  out[((size_t)b * TPRED + s) * NNODES + n] = acc;
}

__global__ void k_init_xprev(const float* __restrict__ xhist, float* __restrict__ xprev) {
  int m = blockIdx.x * 256 + threadIdx.x;
  if (m >= MROWS) return;
  int b = m / NNODES, n = m % NNODES;
  xprev[m] = xhist[((size_t)b * THIST + (THIST - 1)) * NNODES + n];
}

// ---------------- host orchestration ----------------
static inline size_t alignUp(size_t x) { return (x + 255) & ~(size_t)255; }

extern "C" void kernel_launch(void* const* d_in, const int* in_sizes, int n_in,
                              void* d_out, int out_size, void* d_ws, size_t ws_size,
                              hipStream_t stream) {
  (void)in_sizes; (void)n_in; (void)out_size; (void)ws_size;
  const float* x_hist   = (const float*)d_in[0];
  const float* features = (const float*)d_in[1];
  const int*   eidx     = (const int*)d_in[2];
  const float* W_cheb   = (const float*)d_in[3];
  const float* b_cheb   = (const float*)d_in[4];
  const float* W_ih     = (const float*)d_in[5];
  const float* b_ih     = (const float*)d_in[6];
  const float* W_hh     = (const float*)d_in[7];
  const float* b_hh     = (const float*)d_in[8];
  const float* W_fc     = (const float*)d_in[9];
  const float* b_fc     = (const float*)d_in[10];
  float* out = (float*)d_out;

  const int* srcI = eidx;
  const int* dstI = eidx + EDG;

  // ---- workspace carve ----
  char* p = (char*)d_ws;  size_t off = 0;
  auto grab = [&](size_t bytes) { void* r = p + off; off += alignUp(bytes); return r; };
  float*          deg     = (float*)grab(NNODES * 4);
  float*          ew      = (float*)grab(EDG * 4);
  unsigned short* WchebT  = (unsigned short*)grab((size_t)GCNF * CHK * 2);   // [N=256][K=96]
  unsigned short* Wih_bf  = (unsigned short*)grab((size_t)GATES * XGK * 2);  // [N=768][K=288]
  unsigned short* Whh_bf  = (unsigned short*)grab((size_t)GATES * HID * 2);  // [N=768][K=256]
  float*          h       = (float*)grab((size_t)MROWS * HID * 4);
  unsigned short* hbf     = (unsigned short*)grab((size_t)MROWS * HID * 2);
  float*          Tx0     = (float*)grab((size_t)MROWS * IND * 4);
  float*          Tx1     = (float*)grab((size_t)MROWS * IND * 4);
  float*          Tx2     = (float*)grab((size_t)MROWS * IND * 4);
  unsigned short* Acheb   = (unsigned short*)grab((size_t)MROWS * CHK * 2);
  unsigned short* xg      = (unsigned short*)grab((size_t)MROWS * XGK * 2);
  float*          gi      = (float*)grab((size_t)MROWS * GATES * 4);
  float*          gh      = (float*)grab((size_t)MROWS * GATES * 4);
  float*          xprev   = (float*)grab((size_t)MROWS * 4);

  auto cdiv = [](long a, long b) { return (int)((a + b - 1) / b); };

  // ---- one-time setup (recomputed every call for determinism) ----
  (void)hipMemsetAsync(deg, 0, NNODES * 4, stream);
  k_deg<<<cdiv(EDG, 256), 256, 0, stream>>>(srcI, deg);
  k_edgew<<<cdiv(EDG, 256), 256, 0, stream>>>(srcI, dstI, deg, ew);
  // W_cheb is [K=96][N=256] -> transpose to N-major; W_ih/W_hh are already [N][K]
  k_transpose_bf<<<cdiv((long)CHK * GCNF, 256), 256, 0, stream>>>(W_cheb, WchebT, CHK, GCNF);
  k_f2bf<<<cdiv((long)GATES * XGK, 256), 256, 0, stream>>>(W_ih, Wih_bf, GATES * XGK);
  k_f2bf<<<cdiv((long)GATES * HID, 256), 256, 0, stream>>>(W_hh, Whh_bf, GATES * HID);
  (void)hipMemsetAsync(h,   0, (size_t)MROWS * HID * 4, stream);
  (void)hipMemsetAsync(hbf, 0, (size_t)MROWS * HID * 2, stream);

  const dim3 gGemmGates(MROWS / 64, GATES / 64);   // 750 x 12
  const dim3 gGemmCheb (MROWS / 64, GCNF  / 64);   // 750 x 4
  const int  gElem  = cdiv((long)MROWS * IND, 256);
  const int  gScat  = cdiv((long)BB * EDG, 256);
  const int  gGru   = cdiv((long)MROWS * HID, 256);

  auto step = [&](int use_hist, int thist, int tf) {
    k_build_xt<<<gElem, 256, 0, stream>>>(x_hist, xprev, use_hist, thist, features, tf,
                                          Tx0, xg, Acheb);
    (void)hipMemsetAsync(Tx1, 0, (size_t)MROWS * IND * 4, stream);
    k_scatter<<<gScat, 256, 0, stream>>>(Tx0, Tx1, srcI, dstI, ew);
    (void)hipMemsetAsync(Tx2, 0, (size_t)MROWS * IND * 4, stream);
    k_scatter<<<gScat, 256, 0, stream>>>(Tx1, Tx2, srcI, dstI, ew);
    k_combine<<<gElem, 256, 0, stream>>>(Tx0, Tx1, Tx2, Acheb);
    // ChebConv GEMM, fused bias+sigmoid -> bf16 into xg[:, 32:288]
    k_gemm_bf16<1><<<gGemmCheb, 128, 0, stream>>>(Acheb, WchebT, GCNF, CHK, b_cheb,
                                                  nullptr, xg, XGK, IND);
    // gate GEMMs
    k_gemm_bf16<0><<<gGemmGates, 128, 0, stream>>>(xg,  Wih_bf, GATES, XGK, b_ih,
                                                   gi, nullptr, 0, 0);
    k_gemm_bf16<0><<<gGemmGates, 128, 0, stream>>>(hbf, Whh_bf, GATES, HID, b_hh,
                                                   gh, nullptr, 0, 0);
    k_gru<<<gGru, 256, 0, stream>>>(gi, gh, h, hbf);
  };

  // encoder: i = 0..HIST-2 : x_hist[:, i], features[:, i+1]
  for (int i = 0; i < THIST - 1; ++i) step(1, i, i + 1);

  // decoder
  k_init_xprev<<<cdiv(MROWS, 256), 256, 0, stream>>>(x_hist, xprev);
  for (int s = 0; s < TPRED; ++s) {
    step(0, 0, THIST + s);
    k_fc<<<cdiv(MROWS, 256), 256, 0, stream>>>(h, W_fc, b_fc, xprev, out, s);
  }
}
